// QuantizedConv2d_12558484374169
// MI455X (gfx1250) — 8-bit matrix operand order repaired and checked by reading, compiled, not run on hardware
//
#include <hip/hip_runtime.h>
#include <math.h>


typedef __attribute__((ext_vector_type(8))) int v8i;
typedef __attribute__((ext_vector_type(4))) int v4i;

#define ZN    16
#define CIN   128
#define HDIM  56
#define WDIM  56
#define COUT  256
#define HP    58
#define WP    66
#define NKT   18

#define XQ_OFF    0
#define XQ_SIZE   (ZN * HP * WP * CIN)
#define AF_OFF    XQ_SIZE
#define AF_SIZE   (NKT * 16 * 32 * 32)
#define BIAS_OFF  (AF_OFF + AF_SIZE)
#define SCALE_OFF (BIAS_OFF + COUT * 4)

#if defined(__gfx1250__) && __has_builtin(__builtin_amdgcn_global_load_async_to_lds_b128)
#define USE_ASYNC_LDS 1
typedef __attribute__((address_space(1))) v4i gv4i;
typedef __attribute__((address_space(3))) v4i lv4i;
#else
#define USE_ASYNC_LDS 0
#endif

__global__ __launch_bounds__(256) void pack_x_kernel(const int* __restrict__ x,
                                                     const int* __restrict__ p_izp,
                                                     char* __restrict__ xq) {
    int t = blockIdx.x * 256 + threadIdx.x;
    int c  = t & (CIN - 1);
    int r  = t >> 7;
    int ww = r % WP;  r /= WP;
    int hh = r % HP;
    int n  = r / HP;
    int h = hh - 1, w = ww - 1;
    char v = (char)p_izp[0];
    if (h >= 0 && h < HDIM && w >= 0 && w < WDIM)
        v = (char)x[((n * CIN + c) * HDIM + h) * WDIM + w];
    xq[t] = v;
}

__global__ __launch_bounds__(256) void pack_w_kernel(const int* __restrict__ w,
                                                     char* __restrict__ afrag) {
    int t    = blockIdx.x * 256 + threadIdx.x;
    int b    = t & 31;
    int lane = (t >> 5) & 31;
    int mt   = (t >> 10) & 15;
    int kt   = t >> 14;
    int m  = mt * 16 + (lane & 15);
    int d  = b >> 2, bb = b & 3;
    int kLocal = ((d >> 1) << 4) + ((lane >> 4) << 3) + ((d & 1) << 2) + bb;
    int blk = kLocal >> 4;
    int cb  = (blk == 1) ? 2 : ((blk == 2) ? 1 : blk);
    int c   = ((kt & 1) << 6) + ((lane >> 4) << 5) + b;
    int tap = kt >> 1;
    int kh = tap / 3, kw = tap % 3;
    afrag[t] = (char)w[((m * CIN + c) * 3 + kh) * 3 + kw];
}

__global__ __launch_bounds__(256) void pack_s_kernel(const int* __restrict__ w,
                                                     const int* __restrict__ bias,
                                                     const float* __restrict__ wscale,
                                                     const float* __restrict__ p_is,
                                                     const float* __restrict__ p_os,
                                                     const int* __restrict__ p_izp,
                                                     int* __restrict__ bias_eff,
                                                     float* __restrict__ scale_eff) {
    int m = threadIdx.x;
    const int* wm = w + m * CIN * 9;
    int s = 0;
    for (int i = 0; i < CIN * 9; ++i) s += wm[i];
    bias_eff[m]  = bias[m] - p_izp[0] * s;
    scale_eff[m] = p_is[0] * wscale[m] / p_os[0];
}

__global__ __launch_bounds__(256) void conv_wmma_kernel(const char* __restrict__ xq,
                                                        const char* __restrict__ afrag,
                                                        const int* __restrict__ bias_eff,
                                                        const float* __restrict__ scale_eff,
                                                        const int* __restrict__ p_ozp,
                                                        float* __restrict__ out) {
    __shared__ int lds_i[(3 * 18 * CIN) / 4];
    char* lds = (char*)lds_i;

    int bid  = blockIdx.x;
    int owt  = bid & 3;
    int rest = bid >> 2;
    int oh = rest % HDIM;
    int n  = rest / HDIM;
    int ow0 = owt << 4;
    int tid = threadIdx.x;

    for (int i = tid; i < 432; i += 256) {
        int c16 = i & 7;
        int pos = i >> 3;
        int j  = pos % 18;
        int kh = pos / 18;
        const char* src = xq +
            ((size_t)((n * HP + oh + kh) * WP + (ow0 + j)) << 7) + (c16 << 4);
        char* dst = lds + i * 16;
#if USE_ASYNC_LDS
        __builtin_amdgcn_global_load_async_to_lds_b128(
            (gv4i*)(const v4i*)src, (lv4i*)(v4i*)dst, 0, 0);
#else
        *(v4i*)dst = *(const v4i*)src;
#endif
    }
#if USE_ASYNC_LDS
#if __has_builtin(__builtin_amdgcn_s_wait_asynccnt)
    __builtin_amdgcn_s_wait_asynccnt(0);
#else
    asm volatile("s_wait_asynccnt 0x0" ::: "memory");
#endif
#endif
    __syncthreads();

    int lane = tid & 31;
    int wv   = tid >> 5;
    int mt0  = wv * 2, mt1 = mt0 + 1;
    int col  = lane & 15;
    int hihalf = lane >> 4;

    v8i acc0 = {0, 0, 0, 0, 0, 0, 0, 0};
    v8i acc1 = {0, 0, 0, 0, 0, 0, 0, 0};

#pragma unroll
    for (int kt = 0; kt < NKT; ++kt) {
        int tap = kt >> 1;
        int kh = tap / 3, kw = tap % 3;
        int c0 = (kt & 1) << 6;
        int lb = ((kh * 18 + col + kw) << 7) + c0 + (hihalf << 5);
        v8i B = *(const v8i*)(lds + lb);
        v8i A0 = *(const v8i*)(afrag + (((kt * 16 + mt0) << 10) + (lane << 5)));
        v8i A1 = *(const v8i*)(afrag + (((kt * 16 + mt1) << 10) + (lane << 5)));
        acc0 = __builtin_amdgcn_wmma_i32_16x16x64_iu8(true, A0, true, B, acc0, false, false);
        acc1 = __builtin_amdgcn_wmma_i32_16x16x64_iu8(true, A1, true, B, acc1, false, false);
    }

    int ow = ow0 + col;
    int rowAdd = hihalf << 3;
    float ozp = (float)p_ozp[0];
    if (ow < WDIM) {
#pragma unroll
        for (int r = 0; r < 8; ++r) {
            int m0 = mt0 * 16 + rowAdd + r;
            float f0 = fmaf((float)(acc0[r] + bias_eff[m0]), scale_eff[m0], ozp);
            f0 = fminf(fmaxf(rintf(f0), -128.0f), 127.0f);
            out[(((size_t)n * COUT + m0) * HDIM + oh) * WDIM + ow] = f0;
            int m1 = mt1 * 16 + rowAdd + r;
            float f1 = fmaf((float)(acc1[r] + bias_eff[m1]), scale_eff[m1], ozp);
            f1 = fminf(fmaxf(rintf(f1), -128.0f), 127.0f);
            out[(((size_t)n * COUT + m1) * HDIM + oh) * WDIM + ow] = f1;
        }
    }
}

extern "C" void kernel_launch(void* const* d_in, const int* in_sizes, int n_in,
                              void* d_out, int out_size, void* d_ws, size_t ws_size,
                              hipStream_t stream) {
    const int*   x      = (const int*)d_in[0];
    const int*   w      = (const int*)d_in[1];
    const int*   bias   = (const int*)d_in[2];
    const float* wscale = (const float*)d_in[3];
    const float* is     = (const float*)d_in[4];
    const float* os     = (const float*)d_in[5];
    const int*   izp    = (const int*)d_in[6];
    const int*   ozp    = (const int*)d_in[7];

    char*  ws        = (char*)d_ws;
    char*  xq        = ws + XQ_OFF;
    char*  afr       = ws + AF_OFF;
    int*   bias_eff  = (int*)(ws + BIAS_OFF);
    float* scale_eff = (float*)(ws + SCALE_OFF);
    float* out       = (float*)d_out;

    pack_x_kernel<<<XQ_SIZE / 256, 256, 0, stream>>>(x, izp, xq);
    pack_w_kernel<<<AF_SIZE / 256, 256, 0, stream>>>(w, afr);
    pack_s_kernel<<<1, 256, 0, stream>>>(w, bias, wscale, is, os, izp, bias_eff, scale_eff);
    conv_wmma_kernel<<<ZN * HDIM * 4, 256, 0, stream>>>(xq, afr, bias_eff, scale_eff, ozp, out);
}
